// SocialScaleConv4x_5102421148354
// MI455X (gfx1250) — matrix operand order repaired and checked by reading, not compiled, not run on hardware
//
#include <hip/hip_runtime.h>

// ---------------------------------------------------------------------------
// SocialScaleConv4x fused kernel for gfx1250 (MI455X)
//   gathered = concat(take(feat_s, idx_s) for s in 1..4)   [N, 512]
//   h   = relu(gathered @ W.T + b)
//   out = layernorm(h) * gamma + beta        (eps = 1e-6)
//
// Roofline: 52.4 GFLOP vs 412 MB of HBM traffic -> ~18 us at 23.3 TB/s with
// f16 WMMA (f32 accum). W is pre-converted to f16 *and* pre-swizzled into
// fragment-major layout so every B-fragment load in the K-loop is a single
// fully-coalesced 1 KB wave burst from the L2-resident 512 KB weight buffer.
// ---------------------------------------------------------------------------

typedef _Float16 h4   __attribute__((ext_vector_type(4)));
typedef _Float16 v8h  __attribute__((ext_vector_type(8)));
typedef _Float16 v16h __attribute__((ext_vector_type(16)));
typedef float    v8f  __attribute__((ext_vector_type(8)));

#define FOURD   512
#define BM      32          // rows per block (N % 32 == 0 for N=100000)
#define NT      (FOURD/16)  // 32 n-tiles
#define KS      (FOURD/32)  // 16 k-steps
#define LN_EPS  1e-6f

// ---- Kernel 0: convert + swizzle W into fragment-major f16 ----------------
// Whs[ntile][kstep][lane][16] ; lane L supplies B-matrix column
//   n = ntile*16 + (L&15), k = kstep*32 + (L>>4)*16 .. +15  (B[k][n] = W[n][k])
// One thread produces one 16-half fragment (reads 64 B of W, writes 32 B).
__global__ __launch_bounds__(256) void convw_kernel(const float* __restrict__ W,
                                                    _Float16* __restrict__ Whs) {
    int t = blockIdx.x * 256 + threadIdx.x;      // 32*16*32 = 16384 fragments
    if (t >= NT * KS * 32) return;
    int ntile = t >> 9;                          // /(16*32)
    int kstep = (t >> 5) & (KS - 1);
    int lane  = t & 31;
    int n  = ntile * 16 + (lane & 15);
    int k0 = kstep * 32 + (lane >> 4) * 16;
    const float* src = W + (size_t)n * FOURD + k0;   // 16 consecutive fp32
    _Float16*    dst = Whs + (size_t)t * 16;
#pragma unroll
    for (int q = 0; q < 4; ++q) {
        float4 v = ((const float4*)src)[q];
        h4 o;
        o[0] = (_Float16)v.x; o[1] = (_Float16)v.y;
        o[2] = (_Float16)v.z; o[3] = (_Float16)v.w;
        ((h4*)dst)[q] = o;
    }
}

// ---- Kernel 1: fused gather + GEMM (WMMA) + bias + relu + layernorm -------
__global__ __launch_bounds__(256) void fused_kernel(
    const float* __restrict__ f1, const int* __restrict__ i1,
    const float* __restrict__ f2, const int* __restrict__ i2,
    const float* __restrict__ f3, const int* __restrict__ i3,
    const float* __restrict__ f4, const int* __restrict__ i4,
    const _Float16* __restrict__ Whs,
    const float* __restrict__ bias,
    const float* __restrict__ gamma,
    const float* __restrict__ beta,
    float* __restrict__ out, int N)
{
    __shared__ _Float16 As[BM][FOURD];   // 32 KB gathered f16 A-panel
    __shared__ float rowSum[BM];
    __shared__ float rowSq[BM];

    const int tid    = threadIdx.x;
    const int lane   = tid & 31;
    const int wave   = tid >> 5;          // 8 waves; wave w owns cols [64w,64w+64)
    const int laneLo = lane & 15;
    const int laneHi = lane >> 4;         // 0 / 1
    const int rowBase = blockIdx.x * BM;

    if (tid < BM) { rowSum[tid] = 0.0f; rowSq[tid] = 0.0f; }

    // ---- gather 32 rows x 512 cols fp32 -> f16 into LDS -------------------
    // chunk = (row, scale, float4-within-128) : 32*4*32 = 4096 chunks, 16/thread
    for (int c = tid; c < BM * 128; c += 256) {
        int r   = c >> 7;                 // local row
        int rem = c & 127;
        int s   = rem >> 5;               // scale 0..3
        int q   = rem & 31;               // float4 index within the 128-col slab
        int rg  = rowBase + r;
        if (rg >= N) rg = N - 1;          // clamp (never hit for N=100000)
        const float* fp = (s == 0) ? f1 : (s == 1) ? f2 : (s == 2) ? f3 : f4;
        const int*   ip = (s == 0) ? i1 : (s == 1) ? i2 : (s == 2) ? i3 : i4;
        int gidx = ip[rg];
        float4 v = ((const float4*)(fp + (size_t)gidx * 128))[q];
        h4 o;
        o[0] = (_Float16)v.x; o[1] = (_Float16)v.y;
        o[2] = (_Float16)v.z; o[3] = (_Float16)v.w;
        *(h4*)&As[r][s * 128 + q * 4] = o;
    }
    __syncthreads();

    // ---- fragment-major B pointers: wave burst = 32 lanes x 32 B contiguous
    const _Float16* bsrc[4];
    int cols[4];
#pragma unroll
    for (int j = 0; j < 4; ++j) {
        int nt = wave * 4 + j;
        cols[j] = nt * 16 + laneLo;
        bsrc[j] = Whs + ((size_t)nt * KS * 32 + lane) * 16;
    }

    v8f acc[2][4];
#pragma unroll
    for (int mt = 0; mt < 2; ++mt)
#pragma unroll
        for (int j = 0; j < 4; ++j)
#pragma unroll
            for (int i = 0; i < 8; ++i) acc[mt][j][i] = 0.0f;

    // ---- K loop: 16 steps of K=32 -----------------------------------------
    for (int ks = 0; ks < KS; ++ks) {
        const int kk = ks * 32;
        // A k order used here (contiguous, same as Whs B): lanes 0-15: K 0-15;
        //                           lanes 16-31: K 16-31.
        v16h a[2];
#pragma unroll
        for (int mt = 0; mt < 2; ++mt) {
            const _Float16* ap = &As[mt * 16 + laneLo][kk + laneHi * 16];
            v8h lo = *(const v8h*)ap;          // ds_load_b128
            v8h hi = *(const v8h*)(ap + 8);    // ds_load_b128
            a[mt] = __builtin_shufflevector(lo, hi,
                     0, 1, 2, 3, 4, 5, 6, 7, 8, 9, 10, 11, 12, 13, 14, 15);
        }
#pragma unroll
        for (int j = 0; j < 4; ++j) {
            v16h bf = *(const v16h*)(bsrc[j] + (size_t)ks * 32 * 16);
#pragma unroll
            for (int mt = 0; mt < 2; ++mt) {
                acc[mt][j] = __builtin_amdgcn_wmma_f32_16x16x32_f16(
                    false, a[mt], false, bf, (short)0, acc[mt][j], false, false);
            }
        }
    }

    // ---- epilogue: bias + relu, per-row sum/sumsq via shfl + ds_add_f32 ---
#pragma unroll
    for (int mt = 0; mt < 2; ++mt) {
#pragma unroll
        for (int j = 0; j < 4; ++j) {
            float bb = bias[cols[j]];
#pragma unroll
            for (int i = 0; i < 8; ++i) {
                float v = acc[mt][j][i] + bb;
                v = fmaxf(v, 0.0f);
                acc[mt][j][i] = v;
            }
            // C layout: VGPR i, lanes 0-15 -> row mt*16+i; lanes 16-31 -> +8
#pragma unroll
            for (int i = 0; i < 8; ++i) {
                float s  = acc[mt][j][i];
                float s2 = s * s;
#pragma unroll
                for (int m = 1; m < 16; m <<= 1) {
                    s  += __shfl_xor(s,  m, 32);
                    s2 += __shfl_xor(s2, m, 32);
                }
                if (laneLo == 0) {
                    int r = mt * 16 + i + laneHi * 8;
                    atomicAdd(&rowSum[r], s);    // ds_add_f32
                    atomicAdd(&rowSq[r],  s2);
                }
            }
        }
    }
    __syncthreads();

    if (tid < BM) {
        float mu  = rowSum[tid] * (1.0f / (float)FOURD);
        float var = rowSq[tid] * (1.0f / (float)FOURD) - mu * mu;
        rowSum[tid] = mu;
        rowSq[tid]  = __frsqrt_rn(var + LN_EPS);
    }
    __syncthreads();

    // ---- normalize + affine + store (coalesced per 16-lane half) ----------
#pragma unroll
    for (int mt = 0; mt < 2; ++mt) {
#pragma unroll
        for (int j = 0; j < 4; ++j) {
            int col = cols[j];
            float gg = gamma[col];
            float be = beta[col];
#pragma unroll
            for (int i = 0; i < 8; ++i) {
                int r = mt * 16 + i + laneHi * 8;
                int rg = rowBase + r;
                if (rg < N) {
                    float mu = rowSum[r];
                    float rs = rowSq[r];
                    out[(size_t)rg * FOURD + col] =
                        (acc[mt][j][i] - mu) * rs * gg + be;
                }
            }
        }
    }
}

// ---------------------------------------------------------------------------
extern "C" void kernel_launch(void* const* d_in, const int* in_sizes, int n_in,
                              void* d_out, int out_size, void* d_ws, size_t ws_size,
                              hipStream_t stream) {
    // dict order: feat_s1, idx_s1, feat_s2, idx_s2, feat_s3, idx_s3,
    //             feat_s4, idx_s4, W, b, gamma, beta
    const float* f1 = (const float*)d_in[0];
    const int*   i1 = (const int*)  d_in[1];
    const float* f2 = (const float*)d_in[2];
    const int*   i2 = (const int*)  d_in[3];
    const float* f3 = (const float*)d_in[4];
    const int*   i3 = (const int*)  d_in[5];
    const float* f4 = (const float*)d_in[6];
    const int*   i4 = (const int*)  d_in[7];
    const float* W     = (const float*)d_in[8];
    const float* bias  = (const float*)d_in[9];
    const float* gamma = (const float*)d_in[10];
    const float* beta  = (const float*)d_in[11];
    float* out = (float*)d_out;
    const int N = in_sizes[1];                 // idx_s1 length

    _Float16* Whs = (_Float16*)d_ws;           // 512 KB swizzled f16 weights

    convw_kernel<<<(NT * KS * 32 + 255) / 256, 256, 0, stream>>>(W, Whs);

    int nblocks = (N + BM - 1) / BM;           // 3125 for N=100000
    fused_kernel<<<nblocks, 256, 0, stream>>>(
        f1, i1, f2, i2, f3, i3, f4, i4, Whs, bias, gamma, beta, out, N);
}
